// BoundaryConditionedFusion_5059471474755
// MI455X (gfx1250) — matrix operand order repaired and checked by reading, not compiled, not run on hardware
//
#include <hip/hip_runtime.h>
#include <hip/hip_bf16.h>
#include <stdint.h>

typedef __attribute__((ext_vector_type(16))) _Float16 v16h;
typedef __attribute__((ext_vector_type(8)))  _Float16 v8h;
typedef __attribute__((ext_vector_type(8)))  float    v8f;

#define EPSBN 1e-5f
#define QSCALE 0.35355339059327373f  /* 8^-0.5 */

// -------------------------------------------------------------------------
// Kernel 1: fold all BatchNorms into weights/biases (tiny, one-time work)
// -------------------------------------------------------------------------
__global__ void prep_kernel(
    const float* ni_g, const float* ni_b, const float* ni_m, const float* ni_v,   // norm_in (128)
    const float* qkv_w,                                                           // [192,128]
    const float* bp_w, const float* bp_b,                                         // bproj conv [64,9],[64]
    const float* bp_g, const float* bp_be, const float* bp_m, const float* bp_v,  // bproj_bn
    const float* pr_w, const float* pr_b,                                         // proj [64,64],[64]
    const float* n_g, const float* n_be, const float* n_m, const float* n_v,      // norm
    const float* lo_w, const float* lo_b,                                         // local [64,128,9],[64]
    const float* lg, const float* lbe, const float* lm, const float* lv,          // local_bn
    _Float16* Wq, float* qbias, _Float16* W2, float* b2,
    float* bcw, float* bcb, float* lw, float* lb)
{
    int tid  = blockIdx.x * blockDim.x + threadIdx.x;
    int nthr = gridDim.x * blockDim.x;
    // qkv weight folded with input BN scale
    for (int idx = tid; idx < 192 * 128; idx += nthr) {
        int i = idx & 127;
        float s = ni_g[i] * rsqrtf(ni_v[i] + EPSBN);
        Wq[idx] = (_Float16)(qkv_w[idx] * s);
    }
    // qkv bias = W @ (beta - mean*scale)
    for (int o = tid; o < 192; o += nthr) {
        float acc = 0.f;
        for (int i = 0; i < 128; ++i) {
            float s = ni_g[i] * rsqrtf(ni_v[i] + EPSBN);
            acc += qkv_w[o * 128 + i] * (ni_b[i] - ni_m[i] * s);
        }
        qbias[o] = acc;
    }
    // proj folded with output BN
    for (int idx = tid; idx < 64 * 64; idx += nthr) {
        int o = idx >> 6;
        float s = n_g[o] * rsqrtf(n_v[o] + EPSBN);
        W2[idx] = (_Float16)(pr_w[idx] * s);
    }
    for (int o = tid; o < 64; o += nthr) {
        float s  = n_g[o] * rsqrtf(n_v[o] + EPSBN);
        b2[o]    = s * (pr_b[o] - n_m[o]) + n_be[o];
        float bs = bp_g[o] * rsqrtf(bp_v[o] + EPSBN);
        bcb[o]   = bs * (bp_b[o] - bp_m[o]) + bp_be[o];
        float ls = lg[o] * rsqrtf(lv[o] + EPSBN);
        lb[o]    = ls * (lo_b[o] - lm[o]) + lbe[o];
    }
    for (int idx = tid; idx < 64 * 9; idx += nthr) {
        int c = idx / 9;
        float bs = bp_g[c] * rsqrtf(bp_v[c] + EPSBN);
        bcw[idx] = bp_w[idx] * bs;
    }
    for (int idx = tid; idx < 64 * 128 * 9; idx += nthr) {
        int c = idx / (128 * 9);
        float ls = lg[c] * rsqrtf(lv[c] + EPSBN);
        lw[idx] = lo_w[idx] * ls;
    }
}

// -------------------------------------------------------------------------
// Kernel 2: bilinear (antialias triangle) downsample 128x128 -> 64x64
// -------------------------------------------------------------------------
__global__ void bdown_kernel(const float* __restrict__ bm, float* __restrict__ bd)
{
    int t = blockIdx.x * blockDim.x + threadIdx.x;
    if (t >= 2 * 64 * 64) return;
    int x = t & 63, y = (t >> 6) & 63, b = t >> 12;
    const float w4[4] = {0.125f, 0.375f, 0.375f, 0.125f};
    float acc = 0.f, wsum = 0.f;
    for (int iy = 0; iy < 4; ++iy) {
        int sy = 2 * y - 1 + iy;
        if (sy < 0 || sy > 127) continue;
        for (int ix = 0; ix < 4; ++ix) {
            int sx = 2 * x - 1 + ix;
            if (sx < 0 || sx > 127) continue;
            float w = w4[iy] * w4[ix];
            acc  += w * bm[b * 16384 + sy * 128 + sx];
            wsum += w;
        }
    }
    bd[t] = acc / wsum;
}

// -------------------------------------------------------------------------
// Kernel 3: 3x3 conv (1->64 ch) + folded BN + sigmoid -> bw = 1 + sigmoid
// -------------------------------------------------------------------------
__global__ void bweight_kernel(const float* __restrict__ bd, const float* __restrict__ bcw,
                               const float* __restrict__ bcb, float* __restrict__ bw)
{
    int t = blockIdx.x * blockDim.x + threadIdx.x;
    if (t >= 2 * 64 * 64) return;
    int x = t & 63, y = (t >> 6) & 63, b = t >> 12;
    float patch[9];
    for (int dy = -1; dy <= 1; ++dy)
        for (int dx = -1; dx <= 1; ++dx) {
            int yy = y + dy, xx = x + dx;
            patch[(dy + 1) * 3 + (dx + 1)] =
                (yy >= 0 && yy < 64 && xx >= 0 && xx < 64) ? bd[b * 4096 + yy * 64 + xx] : 0.f;
        }
    int n = y * 64 + x;
    for (int c = 0; c < 64; ++c) {
        float s = bcb[c];
        #pragma unroll
        for (int k = 0; k < 9; ++k) s += bcw[c * 9 + k] * patch[k];
        bw[b * 262144 + c * 4096 + n] = 1.f + 1.f / (1.f + __expf(-s));
    }
}

// -------------------------------------------------------------------------
// Kernel 4: pack x = concat(f1,f2) transposed to f16  Xt[bn][128]
// -------------------------------------------------------------------------
__global__ void pack_x_kernel(const float* __restrict__ f1, const float* __restrict__ f2,
                              _Float16* __restrict__ Xt)
{
    int t = blockIdx.x * blockDim.x + threadIdx.x;
    if (t >= 8192 * 128) return;
    int i = t & 127, bn = t >> 7;
    int b = bn >> 12, nn = bn & 4095;
    float v = (i < 64) ? f1[b * 262144 + i * 4096 + nn]
                       : f2[b * 262144 + (i - 64) * 4096 + nn];
    Xt[t] = (_Float16)v;
}

// -------------------------------------------------------------------------
// Kernel 5: local 3x3 conv (128->64) + folded BN
// -------------------------------------------------------------------------
__global__ void local_kernel(const float* __restrict__ f1, const float* __restrict__ f2,
                             const float* __restrict__ lw, const float* __restrict__ lb,
                             float* __restrict__ localbuf)
{
    int t = blockIdx.x * blockDim.x + threadIdx.x;
    if (t >= 2 * 64 * 4096) return;
    int n = t & 4095, c = (t >> 12) & 63, b = t >> 18;
    int x = n & 63, y = n >> 6;
    float s = lb[c];
    for (int ic = 0; ic < 128; ++ic) {
        const float* src = ((ic < 64) ? f1 : f2) + b * 262144 + (ic & 63) * 4096;
        const float* w   = lw + (c * 128 + ic) * 9;
        #pragma unroll
        for (int dy = -1; dy <= 1; ++dy) {
            int yy = y + dy;
            if (yy < 0 || yy > 63) continue;
            #pragma unroll
            for (int dx = -1; dx <= 1; ++dx) {
                int xx = x + dx;
                if (xx < 0 || xx > 63) continue;
                s += w[(dy + 1) * 3 + (dx + 1)] * src[yy * 64 + xx];
            }
        }
    }
    localbuf[b * 262144 + c * 4096 + n] = s;
}

// -------------------------------------------------------------------------
// Kernel 6: qkv GEMM  [192,128]x[128,8192] via v_wmma_f32_16x16x32_f16.
// Epilogue: +bias, * (1+bw) for q/k, write f16 Qt/Kt (transposed) and Vh.
// -------------------------------------------------------------------------
__global__ __launch_bounds__(256) void gemm_qkv_kernel(
    const _Float16* __restrict__ Wq, const float* __restrict__ qbias,
    const _Float16* __restrict__ Xt, const float* __restrict__ bw,
    _Float16* __restrict__ Qt, _Float16* __restrict__ Kt, _Float16* __restrict__ Vh)
{
    int wid = threadIdx.x >> 5, lane = threadIdx.x & 31;
    int tile = blockIdx.x * 8 + wid;          // 0..6143
    int ot = tile / 512, nt = tile % 512;
    int lane16 = lane & 15;
    bool loH = lane < 16;
    int row = ot * 16 + lane16;
    int n   = nt * 16 + lane16;
    v8f acc = {};
    #pragma unroll
    for (int k0 = 0; k0 < 128; k0 += 32) {
        int baseK = k0 + (loH ? 0 : 16);
        v8h a0 = *(const v8h*)(Wq + row * 128 + baseK);
        v8h a1 = *(const v8h*)(Wq + row * 128 + baseK + 8);
        v16h A;
        #pragma unroll
        for (int i = 0; i < 8; ++i) { A[i] = a0[i]; A[i + 8] = a1[i]; }
        v16h Bm = *(const v16h*)(Xt + (size_t)n * 128 + k0 + (loH ? 0 : 16));
        acc = __builtin_amdgcn_wmma_f32_16x16x32_f16(false, A, false, Bm, (short)0, acc, false, false);
    }
    int b = n >> 12, nn = n & 4095;
    int obase = ot * 16 + (loH ? 0 : 8);
    if (ot < 8) {                     // q or k: apply boundary weight
        int cbase = obase & 63;
        v8h pk;
        #pragma unroll
        for (int r = 0; r < 8; ++r) {
            int o = obase + r;
            float v = (acc[r] + qbias[o]) * bw[b * 262144 + (cbase + r) * 4096 + nn];
            pk[r] = (_Float16)((ot < 4) ? v * QSCALE : v);
        }
        if (ot < 4) *(v8h*)(Qt + (size_t)n * 64 + obase)        = pk;
        else        *(v8h*)(Kt + (size_t)n * 64 + (obase - 64)) = pk;
    } else {                          // v: [d, bn] layout, f16
        #pragma unroll
        for (int r = 0; r < 8; ++r) {
            int o = obase + r;
            Vh[(size_t)(o - 128) * 8192 + n] = (_Float16)(acc[r] + qbias[o]);
        }
    }
}

// -------------------------------------------------------------------------
// Kernel 7: flash attention. One wave = one (b, head, 16 q-rows) tile;
// streams 32 keys/step: 2 WMMA (QK^T) + exp + 1 WMMA (PV).
// Softmax denominator computed BY the matrix core: V's B-operand column
// d=8 is all-ones, so Oacc[:,8] accumulates sum(exp(s)) in lockstep.
// P routed via LDS to convert D-layout -> A-layout.
// -------------------------------------------------------------------------
__global__ __launch_bounds__(256) void attn_kernel(
    const _Float16* __restrict__ Qt, const _Float16* __restrict__ Kt,
    const _Float16* __restrict__ Vh, _Float16* __restrict__ AoT)
{
    __shared__ __align__(32) _Float16 Plds[8][16 * 32];
    int wid = threadIdx.x >> 5, lane = threadIdx.x & 31;
    int unit = blockIdx.x * 8 + wid;          // 0..4095
    int qt = unit & 255, h = (unit >> 8) & 7, b = unit >> 11;
    int lane16 = lane & 15;
    bool loH = lane < 16;
    int bnq = b * 4096 + qt * 16;

    v16h Aq = {};
    if (loH) {
        v8h q8 = *(const v8h*)(Qt + (size_t)(bnq + lane16) * 64 + h * 8);
        #pragma unroll
        for (int i = 0; i < 8; ++i) Aq[i] = q8[i];
    }
    v8f Oacc = {};
    _Float16* Pw = &Plds[wid][0];

    for (int kb = 0; kb < 128; ++kb) {
        int kbase = kb * 32;
        int bnk = b * 4096 + kbase;
        v16h Bk0 = {}, Bk1 = {};
        if (loH) {
            v8h k0 = *(const v8h*)(Kt + (size_t)(bnk + lane16) * 64 + h * 8);
            v8h k1 = *(const v8h*)(Kt + (size_t)(bnk + 16 + lane16) * 64 + h * 8);
            #pragma unroll
            for (int i = 0; i < 8; ++i) { Bk0[i] = k0[i]; Bk1[i] = k1[i]; }
        }
        v8f S0 = {}, S1 = {};
        S0 = __builtin_amdgcn_wmma_f32_16x16x32_f16(false, Aq, false, Bk0, (short)0, S0, false, false);
        S1 = __builtin_amdgcn_wmma_f32_16x16x32_f16(false, Aq, false, Bk1, (short)0, S1, false, false);
        // P = exp(S). Scores are small (|s| < ~10) so no max-subtraction is
        // needed; softmax is exactly exp(s)/sum(exp(s)).
        #pragma unroll
        for (int r = 0; r < 8; ++r) {
            int m = r + (loH ? 0 : 8);
            Pw[m * 32 + lane16]      = (_Float16)__expf(S0[r]);
            Pw[m * 32 + 16 + lane16] = (_Float16)__expf(S1[r]);
        }
        asm volatile("s_wait_dscnt 0" ::: "memory");  // P stores visible before A-layout reads
        int off0 = loH ? 0 : 16;
        v8h p0v = *(const v8h*)(Pw + lane16 * 32 + off0);
        v8h p1v = *(const v8h*)(Pw + lane16 * 32 + off0 + 8);
        v16h Ap;
        #pragma unroll
        for (int i = 0; i < 8; ++i) { Ap[i] = p0v[i]; Ap[i + 8] = p1v[i]; }
        v16h Bv = {};
        if (lane16 < 8) {
            Bv = *(const v16h*)(Vh + (size_t)(h * 8 + lane16) * 8192 + bnk + (loH ? 0 : 16));
        } else if (lane16 == 8) {     // ones column -> Oacc[:,8] = softmax denominator
            #pragma unroll
            for (int i = 0; i < 16; ++i) Bv[i] = (_Float16)1.0f;
        }
        Oacc = __builtin_amdgcn_wmma_f32_16x16x32_f16(false, Ap, false, Bv, (short)0, Oacc, false, false);
        if (kb + 1 < 128)
            __builtin_prefetch(Kt + (size_t)(bnk + 32 + lane16) * 64 + h * 8, 0, 0);
    }
    // normalize: denominator for row m lives in column d=8 (lane 8 / lane 24)
    int d = lane16;
    #pragma unroll
    for (int r = 0; r < 8; ++r) {
        float l = __shfl(Oacc[r], 8, 16);
        if (d < 8) {
            int m = r + (loH ? 0 : 8);
            AoT[(size_t)(bnq + m) * 64 + h * 8 + d] = (_Float16)(Oacc[r] / l);
        }
    }
}

// -------------------------------------------------------------------------
// Kernel 8: proj GEMM [64,64]x[64,8192] + bias + local + ReLU -> d_out
// -------------------------------------------------------------------------
__global__ __launch_bounds__(256) void gemm_proj_kernel(
    const _Float16* __restrict__ W2, const float* __restrict__ b2,
    const _Float16* __restrict__ AoT, const float* __restrict__ localbuf,
    float* __restrict__ out)
{
    int wid = threadIdx.x >> 5, lane = threadIdx.x & 31;
    int tile = blockIdx.x * 8 + wid;          // 0..2047
    int ot = tile >> 9, nt = tile & 511;
    int lane16 = lane & 15;
    bool loH = lane < 16;
    int row = ot * 16 + lane16;
    int n   = nt * 16 + lane16;
    v8f acc = {};
    #pragma unroll
    for (int k0 = 0; k0 < 64; k0 += 32) {
        int baseK = k0 + (loH ? 0 : 16);
        v8h a0 = *(const v8h*)(W2 + row * 64 + baseK);
        v8h a1 = *(const v8h*)(W2 + row * 64 + baseK + 8);
        v16h A;
        #pragma unroll
        for (int i = 0; i < 8; ++i) { A[i] = a0[i]; A[i + 8] = a1[i]; }
        v16h Bm = *(const v16h*)(AoT + (size_t)n * 64 + k0 + (loH ? 0 : 16));
        acc = __builtin_amdgcn_wmma_f32_16x16x32_f16(false, A, false, Bm, (short)0, acc, false, false);
    }
    int b = n >> 12, nn = n & 4095;
    int obase = ot * 16 + (loH ? 0 : 8);
    #pragma unroll
    for (int r = 0; r < 8; ++r) {
        int o = obase + r;
        size_t idx = (size_t)b * 262144 + (size_t)o * 4096 + nn;
        float v = acc[r] + b2[o] + localbuf[idx];
        out[idx] = fmaxf(v, 0.f);
    }
}

// -------------------------------------------------------------------------
extern "C" void kernel_launch(void* const* d_in, const int* in_sizes, int n_in,
                              void* d_out, int out_size, void* d_ws, size_t ws_size,
                              hipStream_t stream)
{
    const float* f1   = (const float*)d_in[0];
    const float* f2   = (const float*)d_in[1];
    const float* bm   = (const float*)d_in[2];
    const float* ni_g = (const float*)d_in[3];
    const float* ni_b = (const float*)d_in[4];
    const float* ni_m = (const float*)d_in[5];
    const float* ni_v = (const float*)d_in[6];
    const float* qkvw = (const float*)d_in[7];
    const float* bp_w = (const float*)d_in[8];
    const float* bp_b = (const float*)d_in[9];
    const float* bp_g = (const float*)d_in[10];
    const float* bp_be= (const float*)d_in[11];
    const float* bp_m = (const float*)d_in[12];
    const float* bp_v = (const float*)d_in[13];
    const float* pr_w = (const float*)d_in[14];
    const float* pr_b = (const float*)d_in[15];
    const float* n_g  = (const float*)d_in[16];
    const float* n_be = (const float*)d_in[17];
    const float* n_m  = (const float*)d_in[18];
    const float* n_v  = (const float*)d_in[19];
    const float* lo_w = (const float*)d_in[20];
    const float* lo_b = (const float*)d_in[21];
    const float* lg   = (const float*)d_in[22];
    const float* lbe  = (const float*)d_in[23];
    const float* lm   = (const float*)d_in[24];
    const float* lv   = (const float*)d_in[25];

    char* w = (char*)d_ws;
    size_t off = 0;
    auto take = [&](size_t bytes) -> void* {
        void* p = w + off;
        off = (off + bytes + 255) & ~(size_t)255;
        return p;
    };
    float*     bw       = (float*)    take(2 * 64 * 4096 * 4);   // boundary weight (1+sigmoid)
    _Float16*  Wq       = (_Float16*) take(192 * 128 * 2);
    float*     qbias    = (float*)    take(192 * 4);
    _Float16*  W2       = (_Float16*) take(64 * 64 * 2);
    float*     b2       = (float*)    take(64 * 4);
    float*     bcw      = (float*)    take(64 * 9 * 4);
    float*     bcb      = (float*)    take(64 * 4);
    float*     lw       = (float*)    take(64 * 128 * 9 * 4);
    float*     lb       = (float*)    take(64 * 4);
    float*     bd       = (float*)    take(2 * 4096 * 4);
    _Float16*  Xt       = (_Float16*) take((size_t)8192 * 128 * 2);
    _Float16*  Qt       = (_Float16*) take((size_t)8192 * 64 * 2);
    _Float16*  Kt       = (_Float16*) take((size_t)8192 * 64 * 2);
    _Float16*  Vh       = (_Float16*) take((size_t)64 * 8192 * 2);
    float*     localbuf = (float*)    take((size_t)524288 * 4);
    _Float16*  AoT      = (_Float16*) take((size_t)8192 * 64 * 2);

    prep_kernel<<<64, 256, 0, stream>>>(
        ni_g, ni_b, ni_m, ni_v, qkvw, bp_w, bp_b, bp_g, bp_be, bp_m, bp_v,
        pr_w, pr_b, n_g, n_be, n_m, n_v, lo_w, lo_b, lg, lbe, lm, lv,
        Wq, qbias, W2, b2, bcw, bcb, lw, lb);
    bdown_kernel  <<<32,  256, 0, stream>>>(bm, bd);
    bweight_kernel<<<32,  256, 0, stream>>>(bd, bcw, bcb, bw);
    pack_x_kernel <<<4096, 256, 0, stream>>>(f1, f2, Xt);
    local_kernel  <<<2048, 256, 0, stream>>>(f1, f2, lw, lb, localbuf);
    gemm_qkv_kernel<<<768, 256, 0, stream>>>(Wq, qbias, Xt, bw, Qt, Kt, Vh);
    attn_kernel   <<<512, 256, 0, stream>>>(Qt, Kt, Vh, AoT);
    gemm_proj_kernel<<<256, 256, 0, stream>>>(W2, b2, AoT, localbuf, (float*)d_out);
}
